// MoeActor_42984032698406
// MI455X (gfx1250) — matrix operand order repaired and checked by reading, not compiled, not run on hardware
//
#include <hip/hip_runtime.h>
#include <hip/hip_bf16.h>
#include <math.h>

// Problem dims (fixed by reference)
#define E_   20
#define IN_  512
#define H1_  1024
#define H2_  512
#define H3_  256
#define A_   32
#define B_   4096
#define MT   64        // batch-tile rows per workgroup
#define NTHREADS 256   // 8 waves (wave32)

typedef __attribute__((ext_vector_type(16))) __bf16 v16bf;
typedef __attribute__((ext_vector_type(8)))  float  v8f;

using bf16s = __hip_bfloat16;   // 2-byte storage type

union FragAB { uint4 u[2]; v16bf v; };
union AccU   { v8f v; float f[8]; };

// ---------------------------------------------------------------------------
// One GEMM layer of the MLP, fully in-LDS activations, bf16 WMMA, f32 acc.
//   sIn : LDS bf16 [MT][K], K-contiguous, K % 32 == 0
//   Wt  : global bf16 [N][K] (pre-transposed, zero-padded), N % 32 == 0
//   bias: global f32; valid for n < nReal (else 0)
//   sOut: LDS bf16 [MT][N] (ELU path) or LDS f32 [MT][N] (final layer)
// Wave tiling: 8 waves = 4 (M) x 2 (N); NR=2 accumulator tiles per wave
// when N >= 64 to break WMMA->WMMA RAW chains.
// ---------------------------------------------------------------------------
template<bool ELU, bool OUTF32>
__device__ __forceinline__ void mlp_layer(
    const bf16s* __restrict__ Wt, const float* __restrict__ bias,
    int K, int N, int nReal,
    const bf16s* __restrict__ sIn, void* __restrict__ sOut)
{
    const int tid   = threadIdx.x;
    const int lane  = tid & 31;
    const int wave  = tid >> 5;
    const int waveM = wave >> 1;             // 0..3
    const int waveN = wave & 1;              // 0..1
    const int m0    = waveM * 16;
    const int NR    = (N >= 64) ? 2 : 1;
    const int colsPerIter = 32 * NR;         // 2 N-waves * NR tiles * 16
    const int nIters      = N / colsPerIter;

    const int half  = lane >> 4;             // 0 or 1
    const int l15   = lane & 15;
    const int mA    = m0 + l15;              // A-frag row (M = lane&15)
    const int kAoff = half * 16;             // A-frag: K runs [0..15]/[16..31]
    const int kBoff = half * 16;             // B-frag: K runs [0..15]/[16..31]

    for (int it = 0; it < nIters; ++it) {
        v8f acc[2] = { {}, {} };
        for (int k0 = 0; k0 < K; k0 += 32) {
            // A fragment: two ds_load_b128 per lane (16 bf16)
            FragAB a;
            const uint4* pa = (const uint4*)(sIn + (size_t)mA * K + k0 + kAoff);
            a.u[0] = pa[0];      // K = k0+kAoff .. +7
            a.u[1] = pa[1];      // K = k0+kAoff+8 .. +15
            #pragma unroll
            for (int r = 0; r < 2; ++r) {
                if (r >= NR) break;
                const int n0 = it * colsPerIter + (waveN * NR + r) * 16;
                const int nB = n0 + l15;     // B-frag column (N = lane&15)
                FragAB b;
                const uint4* pb = (const uint4*)(Wt + (size_t)nB * K + k0 + kBoff);
                b.u[0] = pb[0];
                b.u[1] = pb[1];
                acc[r] = __builtin_amdgcn_wmma_f32_16x16x32_bf16(
                    false, a.v, false, b.v, (short)0, acc[r], false, false);
            }
        }
        // Epilogue: bias + ELU, spill C (n = lane&15, m = m0 + half*8 + r8)
        #pragma unroll
        for (int r = 0; r < 2; ++r) {
            if (r >= NR) break;
            const int n0 = it * colsPerIter + (waveN * NR + r) * 16;
            const int n  = n0 + l15;
            const float bv = (n < nReal) ? bias[n] : 0.f;
            AccU c; c.v = acc[r];
            #pragma unroll
            for (int r8 = 0; r8 < 8; ++r8) {
                const int m = m0 + half * 8 + r8;
                float v = c.f[r8] + bv;
                if (ELU) v = (v > 0.f) ? v : (__expf(v) - 1.f);
                if (OUTF32) ((float*)sOut)[(size_t)m * N + n] = v;
                else        ((bf16s*)sOut)[(size_t)m * N + n] = __float2bfloat16(v);
            }
        }
    }
    __syncthreads();
}

__device__ __forceinline__ void stage_tile(const bf16s* __restrict__ g,
                                           bf16s* __restrict__ s, int elems)
{
    const uint4* gs = (const uint4*)g;
    uint4* ss = (uint4*)s;
    for (int i = threadIdx.x; i < elems / 8; i += NTHREADS) ss[i] = gs[i];
    __syncthreads();
}

// ---------------------------------------------------------------------------
// Weight transpose + fp32->bf16 convert: src [e][K][N] -> dst [e][nPad][K]
// (rows n in [N, nPad) zero-filled). Tiled via LDS for coalescing.
// ---------------------------------------------------------------------------
extern "C" __global__ void k_convT(const float* __restrict__ src,
                                   bf16s* __restrict__ dst,
                                   int K, int N, int nPad)
{
    __shared__ float tile[32][33];
    const int e  = blockIdx.z;
    const int n0 = blockIdx.x * 32;
    const int k0 = blockIdx.y * 32;
    src += (size_t)e * K * N;
    dst += (size_t)e * nPad * K;
    for (int i = threadIdx.y; i < 32; i += 8) {
        const int k = k0 + i, n = n0 + threadIdx.x;
        tile[i][threadIdx.x] = (k < K && n < N) ? src[(size_t)k * N + n] : 0.f;
    }
    __syncthreads();
    for (int i = threadIdx.y; i < 32; i += 8) {
        const int n = n0 + i, k = k0 + threadIdx.x;
        if (n < nPad && k < K)
            dst[(size_t)n * K + k] = __float2bfloat16(tile[threadIdx.x][i]);
    }
}

extern "C" __global__ void k_cvt(const float* __restrict__ src,
                                 bf16s* __restrict__ dst, int n)
{
    const int i = blockIdx.x * blockDim.x + threadIdx.x;
    if (i < n) dst[i] = __float2bfloat16(src[i]);
}

// ---------------------------------------------------------------------------
// Gate MLP: one 64-row tile per WG, 4 WMMA layers in LDS, row softmax -> gw
// ---------------------------------------------------------------------------
extern "C" __global__ __launch_bounds__(NTHREADS) void k_gate(
    const bf16s* __restrict__ obsBf,
    const bf16s* __restrict__ W1t, const float* __restrict__ b1,
    const bf16s* __restrict__ W2t, const float* __restrict__ b2,
    const bf16s* __restrict__ W3t, const float* __restrict__ b3,
    const bf16s* __restrict__ W4t, const float* __restrict__ b4,
    float* __restrict__ gw)
{
    extern __shared__ char smem[];
    bf16s* sSmall = (bf16s*)smem;                         // MT*512 bf16 (64KB)
    bf16s* sBig   = (bf16s*)(smem + (size_t)MT * IN_ * 2);// MT*1024 bf16 (128KB)
    const int b0 = blockIdx.x * MT;

    stage_tile(obsBf + (size_t)b0 * IN_, sSmall, MT * IN_);
    mlp_layer<true,  false>(W1t, b1, IN_, H1_, H1_, sSmall, sBig);
    mlp_layer<true,  false>(W2t, b2, H1_, H2_, H2_, sBig, sSmall);
    mlp_layer<true,  false>(W3t, b3, H2_, H3_, H3_, sSmall, sBig);
    float* sLogit = (float*)sSmall;                       // MT x 32 f32
    mlp_layer<false, true >(W4t, b4, H3_, 32, E_, sBig, sLogit);

    if (threadIdx.x < MT) {
        const float* row = sLogit + threadIdx.x * 32;
        float mx = row[0];
        for (int e = 1; e < E_; ++e) mx = fmaxf(mx, row[e]);
        float ex[E_], sum = 0.f;
        for (int e = 0; e < E_; ++e) { ex[e] = __expf(row[e] - mx); sum += ex[e]; }
        const float inv = 1.f / sum;
        for (int e = 0; e < E_; ++e)
            gw[(size_t)(b0 + threadIdx.x) * E_ + e] = ex[e] * inv;
    }
}

// ---------------------------------------------------------------------------
// Expert MLP: grid (B/MT, E); writes raw actions to partial buffer
// ---------------------------------------------------------------------------
extern "C" __global__ __launch_bounds__(NTHREADS) void k_expert(
    const bf16s* __restrict__ eW1t, const float* __restrict__ eb1,
    const bf16s* __restrict__ eW2t, const float* __restrict__ eb2,
    const bf16s* __restrict__ eW3t, const float* __restrict__ eb3,
    const bf16s* __restrict__ eW4t, const float* __restrict__ eb4,
    const bf16s* __restrict__ obsBf, float* __restrict__ part)
{
    extern __shared__ char smem[];
    bf16s* sSmall = (bf16s*)smem;
    bf16s* sBig   = (bf16s*)(smem + (size_t)MT * IN_ * 2);
    const int b0 = blockIdx.x * MT;
    const int e  = blockIdx.y;

    const bf16s* W1t = eW1t + (size_t)e * H1_ * IN_;
    const bf16s* W2t = eW2t + (size_t)e * H2_ * H1_;
    const bf16s* W3t = eW3t + (size_t)e * H3_ * H2_;
    const bf16s* W4t = eW4t + (size_t)e * A_  * H3_;
    const float* bb1 = eb1 + (size_t)e * H1_;
    const float* bb2 = eb2 + (size_t)e * H2_;
    const float* bb3 = eb3 + (size_t)e * H3_;
    const float* bb4 = eb4 + (size_t)e * A_;

    stage_tile(obsBf + (size_t)b0 * IN_, sSmall, MT * IN_);
    mlp_layer<true,  false>(W1t, bb1, IN_, H1_, H1_, sSmall, sBig);
    mlp_layer<true,  false>(W2t, bb2, H1_, H2_, H2_, sBig, sSmall);
    mlp_layer<true,  false>(W3t, bb3, H2_, H3_, H3_, sSmall, sBig);
    float* sAct = (float*)sSmall;                         // MT x 32 f32
    mlp_layer<false, true >(W4t, bb4, H3_, A_, A_, sBig, sAct);

    float* dst = part + ((size_t)e * B_ + b0) * A_;
    for (int i = threadIdx.x; i < MT * A_; i += NTHREADS) dst[i] = sAct[i];
}

// out[b,a] = sum_e gw[b,e] * part[e,b,a]   (deterministic, no atomics)
extern "C" __global__ void k_reduce(const float* __restrict__ gw,
                                    const float* __restrict__ part,
                                    float* __restrict__ out)
{
    const int idx = blockIdx.x * blockDim.x + threadIdx.x;
    if (idx >= B_ * A_) return;
    const int b = idx >> 5, a = idx & 31;
    float s = 0.f;
    #pragma unroll
    for (int e = 0; e < E_; ++e)
        s = fmaf(gw[(size_t)b * E_ + e], part[((size_t)e * B_ + b) * A_ + a], s);
    out[idx] = s;
}

// ---------------------------------------------------------------------------
extern "C" void kernel_launch(void* const* d_in, const int* in_sizes, int n_in,
                              void* d_out, int out_size, void* d_ws, size_t ws_size,
                              hipStream_t stream) {
    (void)in_sizes; (void)n_in; (void)out_size; (void)ws_size;
    const float* obs = (const float*)d_in[0];
    const float* eW1 = (const float*)d_in[1];  const float* eb1 = (const float*)d_in[2];
    const float* eW2 = (const float*)d_in[3];  const float* eb2 = (const float*)d_in[4];
    const float* eW3 = (const float*)d_in[5];  const float* eb3 = (const float*)d_in[6];
    const float* eW4 = (const float*)d_in[7];  const float* eb4 = (const float*)d_in[8];
    const float* gW1 = (const float*)d_in[9];  const float* gb1 = (const float*)d_in[10];
    const float* gW2 = (const float*)d_in[11]; const float* gb2 = (const float*)d_in[12];
    const float* gW3 = (const float*)d_in[13]; const float* gb3 = (const float*)d_in[14];
    const float* gW4 = (const float*)d_in[15]; const float* gb4 = (const float*)d_in[16];

    // Workspace carve (~65 MB): bf16 obs, bf16 transposed weights, gate w, partials
    char* ws = (char*)d_ws; size_t off = 0;
    auto carve = [&](size_t bytes) {
        char* p = ws + off; off += (bytes + 255) & ~(size_t)255; return p;
    };
    bf16s* obsBf = (bf16s*)carve((size_t)B_ * IN_ * 2);
    bf16s* eW1t  = (bf16s*)carve((size_t)E_ * H1_ * IN_ * 2);
    bf16s* eW2t  = (bf16s*)carve((size_t)E_ * H2_ * H1_ * 2);
    bf16s* eW3t  = (bf16s*)carve((size_t)E_ * H3_ * H2_ * 2);
    bf16s* eW4t  = (bf16s*)carve((size_t)E_ * A_  * H3_ * 2);
    bf16s* gW1t  = (bf16s*)carve((size_t)H1_ * IN_ * 2);
    bf16s* gW2t  = (bf16s*)carve((size_t)H2_ * H1_ * 2);
    bf16s* gW3t  = (bf16s*)carve((size_t)H3_ * H2_ * 2);
    bf16s* gW4t  = (bf16s*)carve((size_t)32  * H3_ * 2);   // 20 -> 32 zero-pad
    float* gw    = (float*)carve((size_t)B_ * E_ * 4);
    float* part  = (float*)carve((size_t)E_ * B_ * A_ * 4);

    const dim3 tb(32, 8, 1);
    k_convT<<<dim3(H1_/32, IN_/32, E_), tb, 0, stream>>>(eW1, eW1t, IN_, H1_, H1_);
    k_convT<<<dim3(H2_/32, H1_/32, E_), tb, 0, stream>>>(eW2, eW2t, H1_, H2_, H2_);
    k_convT<<<dim3(H3_/32, H2_/32, E_), tb, 0, stream>>>(eW3, eW3t, H2_, H3_, H3_);
    k_convT<<<dim3(A_/32,  H3_/32, E_), tb, 0, stream>>>(eW4, eW4t, H3_, A_,  A_);
    k_convT<<<dim3(H1_/32, IN_/32, 1),  tb, 0, stream>>>(gW1, gW1t, IN_, H1_, H1_);
    k_convT<<<dim3(H2_/32, H1_/32, 1),  tb, 0, stream>>>(gW2, gW2t, H1_, H2_, H2_);
    k_convT<<<dim3(H3_/32, H2_/32, 1),  tb, 0, stream>>>(gW3, gW3t, H2_, H3_, H3_);
    k_convT<<<dim3(1,      H3_/32, 1),  tb, 0, stream>>>(gW4, gW4t, H3_, E_,  32);
    k_cvt<<<(B_*IN_ + 255)/256, 256, 0, stream>>>(obs, obsBf, B_ * IN_);

    const size_t lds = (size_t)MT * IN_ * 2 + (size_t)MT * H1_ * 2;  // 192 KB
    k_gate<<<B_/MT, NTHREADS, lds, stream>>>(obsBf, gW1t, gb1, gW2t, gb2,
                                             gW3t, gb3, gW4t, gb4, gw);
    k_expert<<<dim3(B_/MT, E_), NTHREADS, lds, stream>>>(eW1t, eb1, eW2t, eb2,
                                                         eW3t, eb3, eW4t, eb4,
                                                         obsBf, part);
    k_reduce<<<(B_*A_ + 255)/256, 256, 0, stream>>>(gw, part, (float*)d_out);
}
